// Align_54279796687162
// MI455X (gfx1250) — matrix operand order repaired and checked by reading, not compiled, not run on hardware
//
#include <hip/hip_runtime.h>

typedef __attribute__((ext_vector_type(16))) __bf16 v16bf;
typedef __attribute__((ext_vector_type(8)))  float  v8f;

union FragA { v16bf v; unsigned u[8]; };
union FragC { v8f   v; float    f[8]; };

__device__ __forceinline__ unsigned short f2bf(float f) {
  union { float f; unsigned u; } x; x.f = f;
  unsigned r = x.u + 0x7FFFu + ((x.u >> 16) & 1u);
  return (unsigned short)(r >> 16);
}

// ---------------------------------------------------------------------------
// Generic bf16 WMMA GEMM: C[M,N] = A[M,K] * Bt[N,K]^T + bias, optional ReLU.
// Tiles: M=64, N=NT (128 or 256), K-step=32. 512 threads = 16 waves (4M x 4N),
// each wave owns a 16 x (NT/4) slab -> low per-thread VGPR pressure so the
// double-buffered LDS pipeline doesn't spill.
// ---------------------------------------------------------------------------
template <int NT>
__global__ __launch_bounds__(512, 1) __attribute__((amdgpu_waves_per_eu(1)))
void gemm_bf16_nt(
    const unsigned short* __restrict__ A,
    const unsigned short* __restrict__ Bt,
    const float* __restrict__ bias,
    float* __restrict__ C,
    int M, int N, int K, int relu)
{
  constexpr int NSUB = NT / 64;    // 16-wide subtiles per N-wave (2 or 4)
  constexpr int BV   = NT / 128;   // uint4 staged per thread for B (1 or 2)
  __shared__ unsigned short ldsA[2][64 * 32];
  __shared__ unsigned short ldsB[2][NT * 32];

  const int t = threadIdx.x;
  const int lane = t & 31, wv = t >> 5;
  const int mw = wv >> 2, nw = wv & 3;
  const int lm = lane & 15, hi = lane >> 4;
  const int gm0 = blockIdx.x * 64;
  const int gn0 = blockIdx.y * NT;

  int kbA[8];
#pragma unroll
  for (int v = 0; v < 8; ++v)
    kbA[v] = (v < 4) ? (2 * v + 8 * hi) : (16 + 2 * (v - 4) + 8 * hi);

  FragC acc[NSUB];
#pragma unroll
  for (int s = 0; s < NSUB; ++s)
#pragma unroll
    for (int r = 0; r < 8; ++r) acc[s].f[r] = 0.0f;

  const int ar = t >> 3, ako = (t & 7) * 4;   // A stage: 8 thr/row, 4 bf16 each

  uint2 ra;
  uint4 rb[BV];

  auto loadTiles = [&](int k0) {
    ra = make_uint2(0u, 0u);
    if (gm0 + ar < M) ra = *(const uint2*)(A + (size_t)(gm0 + ar) * K + k0 + ako);
#pragma unroll
    for (int i = 0; i < BV; ++i) {
      const int e = t + i * 512;
      const int n = e >> 2, ko = (e & 3) * 8;
      rb[i] = make_uint4(0u, 0u, 0u, 0u);
      if (gn0 + n < N) rb[i] = *(const uint4*)(Bt + (size_t)(gn0 + n) * K + k0 + ko);
    }
  };
  auto storeTiles = [&](int buf) {
    *(uint2*)(&ldsA[buf][ar * 32 + ako]) = ra;
#pragma unroll
    for (int i = 0; i < BV; ++i) {
      const int e = t + i * 512;
      const int n = e >> 2, ko = (e & 3) * 8;
      *(uint4*)(&ldsB[buf][n * 32 + ko]) = rb[i];
    }
  };

  loadTiles(0);
  storeTiles(0);
  __syncthreads();
  int cur = 0;

  for (int k0 = 0; k0 < K; k0 += 32) {
    const bool hasNext = (k0 + 32) < K;
    if (hasNext) loadTiles(k0 + 32);
    if (k0 + 64 < K && gm0 + ar < M)
      __builtin_prefetch(A + (size_t)(gm0 + ar) * K + k0 + 64 + ako, 0, 3);

    const unsigned* A32 = (const unsigned*)ldsA[cur];
    const unsigned* B32 = (const unsigned*)ldsB[cur];
    FragA af;
#pragma unroll
    for (int v = 0; v < 8; ++v)
      af.u[v] = A32[(mw * 16 + lm) * 16 + (kbA[v] >> 1)];
#pragma unroll
    for (int s = 0; s < NSUB; ++s) {
      FragA bf;
      const int n = nw * (NT / 4) + s * 16 + lm;
#pragma unroll
      for (int v = 0; v < 8; ++v)
        bf.u[v] = B32[n * 16 + (kbA[v] >> 1)];
      acc[s].v = __builtin_amdgcn_wmma_f32_16x16x32_bf16(
          false, af.v, false, bf.v, (short)0, acc[s].v, false, false);
    }

    if (hasNext) {
      storeTiles(cur ^ 1);
      __syncthreads();
      cur ^= 1;
    }
  }

#pragma unroll
  for (int s = 0; s < NSUB; ++s) {
    const int gn = gn0 + nw * (NT / 4) + s * 16 + lm;
    if (gn >= N) continue;
    const float bv = bias ? bias[gn] : 0.0f;
#pragma unroll
    for (int r = 0; r < 8; ++r) {
      const int gm = gm0 + mw * 16 + hi * 8 + r;
      if (gm < M) {
        float val = acc[s].f[r] + bv;
        if (relu) val = fmaxf(val, 0.0f);
        C[(size_t)gm * N + gn] = val;
      }
    }
  }
}

// ---------------------------------------------------------------------------
// conv3x3 (Cin=128 -> Cout=256) as implicit-im2col WMMA GEMM over NHWC bf16.
// M-tile = 64 pixels of one image row; N-tile = 256 (all Cout); K = 9*128 in
// 36 steps (pos, ci-block). 512 threads = 16 waves (4M x 4N), double-buffered.
// ---------------------------------------------------------------------------
__global__ __launch_bounds__(512, 1) __attribute__((amdgpu_waves_per_eu(1)))
void conv3x3_wmma_kernel(
    const unsigned short* __restrict__ xb,   // [B,128,128,128] bf16
    const unsigned short* __restrict__ wc,   // [256][1152] bf16 (k=(ky*3+kx)*128+ci)
    const float* __restrict__ bias,
    float* __restrict__ outf,                // [131072,256] f32
    unsigned short* __restrict__ outb)       // [131072,256] bf16
{
  __shared__ unsigned short ldsA[2][64 * 32];
  __shared__ unsigned short ldsB[2][256 * 32];

  const int t = threadIdx.x;
  const int lane = t & 31, wv = t >> 5;
  const int mw = wv >> 2, nw = wv & 3;
  const int lm = lane & 15, hi = lane >> 4;

  const int mb = blockIdx.x;
  const int x0 = (mb & 1) * 64;
  const int y  = (mb >> 1) & 127;
  const int b  = mb >> 8;
  const int pbase = (b * 128 + y) * 128 + x0;

  int kbA[8];
#pragma unroll
  for (int v = 0; v < 8; ++v)
    kbA[v] = (v < 4) ? (2 * v + 8 * hi) : (16 + 2 * (v - 4) + 8 * hi);

  FragC acc[4];
#pragma unroll
  for (int s = 0; s < 4; ++s)
#pragma unroll
    for (int r = 0; r < 8; ++r) acc[s].f[r] = 0.0f;

  const int ai = t >> 3, ako = (t & 7) * 4;

  uint2 ra;
  uint4 rb[2];

  auto loadTiles = [&](int s36) {
    const int pos = s36 >> 2, cib = s36 & 3;
    const int ky = pos / 3;            // arithmetic, keeps out of scratch
    const int kx = pos - ky * 3;
    const int px = x0 + ai + kx - 1;
    const int py = y + ky - 1;
    ra = make_uint2(0u, 0u);
    if (px >= 0 && px < 128 && py >= 0 && py < 128)
      ra = *(const uint2*)(xb + ((size_t)((b * 128 + py) * 128 + px)) * 128
                              + cib * 32 + ako);
#pragma unroll
    for (int i = 0; i < 2; ++i) {
      const int e = t + i * 512;
      const int n = e >> 2, ko = (e & 3) * 8;
      rb[i] = *(const uint4*)(wc + (size_t)n * 1152 + pos * 128 + cib * 32 + ko);
    }
  };
  auto storeTiles = [&](int buf) {
    *(uint2*)(&ldsA[buf][ai * 32 + ako]) = ra;
#pragma unroll
    for (int i = 0; i < 2; ++i) {
      const int e = t + i * 512;
      const int n = e >> 2, ko = (e & 3) * 8;
      *(uint4*)(&ldsB[buf][n * 32 + ko]) = rb[i];
    }
  };

  loadTiles(0);
  storeTiles(0);
  __syncthreads();
  int cur = 0;

  for (int s36 = 0; s36 < 36; ++s36) {
    const bool hasNext = s36 + 1 < 36;
    if (hasNext) loadTiles(s36 + 1);

    const unsigned* A32 = (const unsigned*)ldsA[cur];
    const unsigned* B32 = (const unsigned*)ldsB[cur];
    FragA af;
#pragma unroll
    for (int v = 0; v < 8; ++v)
      af.u[v] = A32[(mw * 16 + lm) * 16 + (kbA[v] >> 1)];
#pragma unroll
    for (int s = 0; s < 4; ++s) {
      FragA bf;
      const int n = nw * 64 + s * 16 + lm;
#pragma unroll
      for (int v = 0; v < 8; ++v)
        bf.u[v] = B32[n * 16 + (kbA[v] >> 1)];
      acc[s].v = __builtin_amdgcn_wmma_f32_16x16x32_bf16(
          false, af.v, false, bf.v, (short)0, acc[s].v, false, false);
    }

    if (hasNext) {
      storeTiles(cur ^ 1);
      __syncthreads();
      cur ^= 1;
    }
  }

#pragma unroll
  for (int s = 0; s < 4; ++s) {
    const int gn = nw * 64 + s * 16 + lm;
    const float bv = bias[gn];
#pragma unroll
    for (int r = 0; r < 8; ++r) {
      const int gm = pbase + mw * 16 + hi * 8 + r;
      float val = fmaxf(acc[s].f[r] + bv, 0.0f);
      outf[(size_t)gm * 256 + gn] = val;
      outb[(size_t)gm * 256 + gn] = f2bf(val);
    }
  }
}

// ---------------------------------------------------------------------------
// NCHW f32 -> NHWC bf16 transpose (32x32 tile in LDS)
// ---------------------------------------------------------------------------
__global__ __launch_bounds__(256) void pack_x_kernel(
    const float* __restrict__ x, unsigned short* __restrict__ xb)
{
  __shared__ float tile[32][33];
  const int bh = blockIdx.x;          // b*128 + h
  const int b = bh >> 7, h = bh & 127;
  const int ci0 = blockIdx.y * 32, w0 = blockIdx.z * 32;
  const int ty = threadIdx.x >> 5, tx = threadIdx.x & 31;
#pragma unroll
  for (int rr = 0; rr < 32; rr += 8)
    tile[ty + rr][tx] = x[((size_t)(b * 128 + ci0 + ty + rr)) * 16384 + h * 128 + w0 + tx];
  __syncthreads();
#pragma unroll
  for (int rr = 0; rr < 32; rr += 8)
    xb[((size_t)((b * 128 + h) * 128 + w0 + ty + rr)) * 128 + ci0 + tx] =
        f2bf(tile[tx][ty + rr]);
}

__global__ void pack_w33_kernel(const float* __restrict__ w, unsigned short* __restrict__ out)
{
  int idx = blockIdx.x * 256 + threadIdx.x;
  if (idx >= 256 * 1152) return;
  int k = idx % 1152, n = idx / 1152;
  int pos = k >> 7, ci = k & 127;
  int ky = pos / 3, kx = pos - ky * 3;
  out[idx] = f2bf(w[((size_t)(n * 128 + ci) * 3 + ky) * 3 + kx]);
}

__global__ void cvt_bf16_kernel(const float* __restrict__ in, unsigned short* __restrict__ out, int n)
{
  int i = blockIdx.x * 256 + threadIdx.x;
  if (i < n) out[i] = f2bf(in[i]);
}

__global__ void copy_f32_kernel(const float* __restrict__ in, float* __restrict__ out, int n)
{
  int i = blockIdx.x * 256 + threadIdx.x;
  if (i < n) out[i] = in[i];
}

// ---------------------------------------------------------------------------
// CCAM: energy[b,c,k] = sum_n Xb[b,n,c] * Yf[b,n,k]
// ---------------------------------------------------------------------------
__global__ __launch_bounds__(256) void energy_kernel(
    const float* __restrict__ Xb, const float* __restrict__ Yf, float* __restrict__ E)
{
  __shared__ float Xs[64 * 16];
  __shared__ float Ys[64 * 16];
  const int b = blockIdx.x >> 4, c0 = (blockIdx.x & 15) * 16;
  const int t = threadIdx.x;
  const int cl = t >> 4, kk = t & 15;
  float acc = 0.0f;
  for (int n0 = 0; n0 < 16384; n0 += 64) {
    __syncthreads();
#pragma unroll
    for (int j = 0; j < 4; ++j) {
      int e = t + j * 256;
      int nl = e >> 4, q = e & 15;
      Xs[e] = Xb[((size_t)(b * 16384 + n0 + nl)) * 256 + c0 + q];
      Ys[e] = Yf[((size_t)(b * 16384 + n0 + nl)) * 16 + q];
    }
    __syncthreads();
#pragma unroll 8
    for (int nl = 0; nl < 64; ++nl)
      acc += Xs[nl * 16 + cl] * Ys[nl * 16 + kk];
  }
  E[((size_t)(b * 256 + c0 + cl)) * 16 + kk] = acc;
}

__global__ void ccam_softmax_kernel(const float* __restrict__ E, float* __restrict__ A)
{
  int r = blockIdx.x * 256 + threadIdx.x;
  if (r >= 2048) return;
  float e[16], mx = -1e30f;
#pragma unroll
  for (int k = 0; k < 16; ++k) { e[k] = E[r * 16 + k]; mx = fmaxf(mx, e[k]); }
  float sm = -1e30f;
#pragma unroll
  for (int k = 0; k < 16; ++k) { e[k] = mx - e[k]; sm = fmaxf(sm, e[k]); }
  float sum = 0.0f;
#pragma unroll
  for (int k = 0; k < 16; ++k) { e[k] = __expf(e[k] - sm); sum += e[k]; }
  float inv = 1.0f / sum;
#pragma unroll
  for (int k = 0; k < 16; ++k) A[r * 16 + k] = e[k] * inv;
}

__global__ __launch_bounds__(256) void xf_kernel(
    const float* __restrict__ Xb, const float* __restrict__ Yf,
    const float* __restrict__ Attn, const float* __restrict__ scale,
    unsigned short* __restrict__ xf)
{
  __shared__ float Yl[16];
  const int p = blockIdx.x;
  const int b = p >> 14;
  const int c = threadIdx.x;
  if (c < 16) Yl[c] = Yf[(size_t)p * 16 + c];
  __syncthreads();
  const float* ar = Attn + ((size_t)(b * 256 + c)) * 16;
  float s = 0.0f;
#pragma unroll
  for (int k = 0; k < 16; ++k) s += ar[k] * Yl[k];
  float val = Xb[(size_t)p * 256 + c] + scale[0] * s;
  xf[(size_t)p * 256 + c] = f2bf(val);
}

// ---------------------------------------------------------------------------
// Shunted mean + linear-interp positional add.
// out[b][cc][ch][pos], row_mode: mean over w=a*4+cc; col: mean over h=cc*32+a.
// ---------------------------------------------------------------------------
__global__ void shunt_kernel(const float* __restrict__ qkv, const float* __restrict__ pos,
                             float* __restrict__ out, int chan_off, int nch,
                             int row_mode, int total)
{
  int idx = blockIdx.x * 256 + threadIdx.x;
  if (idx >= total) return;
  int pi = idx & 127; int tmp = idx >> 7;
  int ch = tmp % nch; tmp /= nch;
  int cc = tmp & 3, b = tmp >> 2;
  float s = 0.0f;
  if (row_mode) {
    for (int a = 0; a < 32; ++a)
      s += qkv[((size_t)(b * 16384 + pi * 128 + a * 4 + cc)) * 512 + chan_off + ch];
  } else {
    for (int a = 0; a < 32; ++a)
      s += qkv[((size_t)(b * 16384 + (cc * 32 + a) * 128 + pi)) * 512 + chan_off + ch];
  }
  s *= (1.0f / 32.0f);
  if (pos) {
    float src = (pi + 0.5f) * 0.125f - 0.5f;
    src = fminf(fmaxf(src, 0.0f), 15.0f);
    int i0 = (int)floorf(src);
    int i1 = i0 + 1; if (i1 > 15) i1 = 15;
    float w = src - (float)i0;
    const float* pr = pos + ((size_t)cc * nch + ch) * 16;
    s += pr[i0] * (1.0f - w) + pr[i1] * w;
  }
  out[idx] = s;
}

// ---------------------------------------------------------------------------
// Axial attention (one head, one chunk per block), online softmax, ReLU,
// bf16 output as [pixel = b*512 + cc*128 + i][channel = g*32 + d].
// ---------------------------------------------------------------------------
__global__ __launch_bounds__(128) void axial_attn_kernel(
    const float* __restrict__ qs, const float* __restrict__ ks,
    const float* __restrict__ vs, unsigned short* __restrict__ outA)
{
  __shared__ float Kt[16 * 128];
  __shared__ float Vt[128 * 32];
  const int blk = blockIdx.x;
  const int g = blk & 7, bc = blk >> 3;       // bc = b*4 + cc
  const int i = threadIdx.x;
  const float* qsb = qs + (size_t)bc * 128 * 128;
  const float* ksb = ks + (size_t)bc * 128 * 128;
  const float* vsb = vs + (size_t)bc * 256 * 128;
  for (int e = i; e < 16 * 128; e += 128)
    Kt[e] = ksb[(g * 16 + (e >> 7)) * 128 + (e & 127)];
  for (int e = i; e < 128 * 32; e += 128) {
    int j = e >> 5, d = e & 31;
    Vt[e] = vsb[(g * 32 + d) * 128 + j];
  }
  float q[16];
#pragma unroll
  for (int kd = 0; kd < 16; ++kd) q[kd] = qsb[(g * 16 + kd) * 128 + i];
  __syncthreads();

  float m = -1e30f, l = 0.0f;
  float accv[32];
#pragma unroll
  for (int d = 0; d < 32; ++d) accv[d] = 0.0f;
  for (int j = 0; j < 128; ++j) {
    float s = 0.0f;
#pragma unroll
    for (int kd = 0; kd < 16; ++kd) s += q[kd] * Kt[kd * 128 + j];
    s *= 0.25f;  // 16^-0.5
    float mn = fmaxf(m, s);
    float corr = __expf(m - mn);
    float p = __expf(s - mn);
    l = l * corr + p;
#pragma unroll
    for (int d = 0; d < 32; ++d)
      accv[d] = accv[d] * corr + p * Vt[j * 32 + d];
    m = mn;
  }
  float inv = 1.0f / l;
  const int b = bc >> 2, cc = bc & 3;
#pragma unroll
  for (int d = 0; d < 32; ++d) {
    float r = fmaxf(accv[d] * inv, 0.0f);
    outA[((size_t)(b * 512 + cc * 128 + i)) * 256 + g * 32 + d] = f2bf(r);
  }
}

// ---------------------------------------------------------------------------
// Depthwise 3x3 (512 groups) + ReLU -> bf16 NHWC
// ---------------------------------------------------------------------------
__global__ void dwconv_kernel(const float* __restrict__ qkv, const float* __restrict__ w,
                              const float* __restrict__ bias, unsigned short* __restrict__ out)
{
  size_t idx = (size_t)blockIdx.x * 256 + threadIdx.x;
  int ch = (int)(idx & 511);
  size_t p = idx >> 9;
  int b = (int)(p >> 14); int hw = (int)(p & 16383);
  int h = hw >> 7, wq = hw & 127;
  float acc = bias[ch];
#pragma unroll
  for (int ky = 0; ky < 3; ++ky)
#pragma unroll
    for (int kx = 0; kx < 3; ++kx) {
      int y = h + ky - 1, xx = wq + kx - 1;
      if (y >= 0 && y < 128 && xx >= 0 && xx < 128)
        acc += w[ch * 9 + ky * 3 + kx] *
               qkv[((size_t)(b * 16384 + y * 128 + xx)) * 512 + ch];
    }
  out[idx] = f2bf(fmaxf(acc, 0.0f));
}

// xx = relu(v + xrow_bcast + xcol_bcast) -> bf16 NHWC
__global__ void xx_kernel(const float* __restrict__ qkv, const float* __restrict__ xr,
                          const float* __restrict__ xc, unsigned short* __restrict__ out)
{
  size_t idx = (size_t)blockIdx.x * 256 + threadIdx.x;
  int ch = (int)(idx & 255);
  size_t p = idx >> 8;
  int b = (int)(p >> 14); int hw = (int)(p & 16383);
  int h = hw >> 7, w = hw & 127;
  int idx2 = h >> 3;
  int a = (h & 7) * 4 + (w >> 5);
  int bc = w & 31;
  float v  = qkv[p * 512 + 256 + ch];
  float vr = xr[((size_t)(b * 512 + idx2 * 32 + a)) * 256 + ch];
  float vc = xc[((size_t)(b * 512 + idx2 * 32 + bc)) * 256 + ch];
  out[idx] = f2bf(fmaxf(v + vr + vc, 0.0f));
}

// out(NCHW) = hsigmoid(proj) * pw
__global__ void final_kernel(const float* __restrict__ proj, const float* __restrict__ pw,
                             float* __restrict__ out)
{
  size_t idx = (size_t)blockIdx.x * 256 + threadIdx.x;
  int w = (int)(idx & 127);
  int h = (int)((idx >> 7) & 127);
  int ch = (int)((idx >> 14) & 255);
  int b = (int)(idx >> 22);
  size_t p = (size_t)(b * 16384 + h * 128 + w);
  float xv = proj[p * 256 + ch];
  float att = fminf(fmaxf(xv + 3.0f, 0.0f), 6.0f) * (1.0f / 6.0f);
  out[idx] = att * pw[p * 256 + ch];
}

// ---------------------------------------------------------------------------
extern "C" void kernel_launch(void* const* d_in, const int* in_sizes, int n_in,
                              void* d_out, int out_size, void* d_ws, size_t ws_size,
                              hipStream_t stream)
{
  (void)in_sizes; (void)n_in; (void)out_size; (void)ws_size;
  const float* x      = (const float*)d_in[0];
  const float* w_ccam = (const float*)d_in[1];
  const float* b_ccam = (const float*)d_in[2];
  const float* w_enc  = (const float*)d_in[3];
  const float* b_enc  = (const float*)d_in[4];
  const float* scale  = (const float*)d_in[5];
  const float* w_q    = (const float*)d_in[6];
  const float* b_q    = (const float*)d_in[7];
  const float* w_k    = (const float*)d_in[8];
  const float* b_k    = (const float*)d_in[9];
  const float* w_v    = (const float*)d_in[10];
  const float* b_v    = (const float*)d_in[11];
  const float* w_dw   = (const float*)d_in[12];
  const float* b_dw   = (const float*)d_in[13];
  const float* w_pw   = (const float*)d_in[14];
  const float* b_pw   = (const float*)d_in[15];
  const float* pos_rq = (const float*)d_in[16];
  const float* pos_rk = (const float*)d_in[17];
  const float* pos_cq = (const float*)d_in[18];
  const float* pos_ck = (const float*)d_in[19];
  const float* w_row  = (const float*)d_in[20];
  const float* b_row  = (const float*)d_in[21];
  const float* w_col  = (const float*)d_in[22];
  const float* b_col  = (const float*)d_in[23];
  const float* w_proj = (const float*)d_in[24];
  const float* b_proj = (const float*)d_in[25];
  float* out = (float*)d_out;

  size_t off = 0;
  auto alloc = [&](size_t bytes) -> char* {
    char* p = (char*)d_ws + off;
    off += (bytes + 255) & ~(size_t)255;
    return p;
  };
  unsigned short* xb      = (unsigned short*)alloc(16777216ull * 2);  // x NHWC bf16
  unsigned short* wc      = (unsigned short*)alloc(294912ull * 2);    // conv3x3 w [256][1152]
  unsigned short* wqkv    = (unsigned short*)alloc(131072ull * 2);    // [512][256]
  unsigned short* wencb   = (unsigned short*)alloc(4096ull * 2);      // [16][256]
  unsigned short* wpwb    = (unsigned short*)alloc(131072ull * 2);    // [256][512]
  unsigned short* wrowb   = (unsigned short*)alloc(65536ull * 2);
  unsigned short* wcolb   = (unsigned short*)alloc(65536ull * 2);
  unsigned short* wprojb  = (unsigned short*)alloc(65536ull * 2);
  float* bqkv             = (float*)alloc(512ull * 4);
  float* ccam_b           = (float*)alloc(33554432ull * 4);           // f32 NHWC; later dw_act
  unsigned short* ccam_bf = (unsigned short*)alloc(33554432ull * 2);
  float* ccam_f           = (float*)alloc(2097152ull * 4);            // [131072,16]
  float* energy           = (float*)alloc(32768ull * 4);
  float* attnb            = (float*)alloc(32768ull * 4);
  unsigned short* xfb     = (unsigned short*)alloc(33554432ull * 2);  // later xx_act
  float* qkvb             = (float*)alloc(67108864ull * 4);           // [131072,512]; later proj
  float* qs_r = (float*)alloc(524288ull * 4);
  float* ks_r = (float*)alloc(524288ull * 4);
  float* vs_r = (float*)alloc(1048576ull * 4);
  float* qs_c = (float*)alloc(524288ull * 4);
  float* ks_c = (float*)alloc(524288ull * 4);
  float* vs_c = (float*)alloc(1048576ull * 4);
  unsigned short* xrow_a = (unsigned short*)alloc(1048576ull * 2);
  unsigned short* xcol_a = (unsigned short*)alloc(1048576ull * 2);
  float* xrow_o = (float*)alloc(1048576ull * 4);
  float* xcol_o = (float*)alloc(1048576ull * 4);
  float* pw_o   = (float*)alloc(33554432ull * 4);
  unsigned short* dw_a = (unsigned short*)ccam_b;  // alias (ccam_b dead after xf)
  unsigned short* xx_a = xfb;                      // alias (xf dead after qkv gemm)
  float* proj_o = qkvb;                            // alias (qkv dead after xx)

  // ---- packing ----
  pack_x_kernel<<<dim3(1024, 4, 4), 256, 0, stream>>>(x, xb);
  pack_w33_kernel<<<1152, 256, 0, stream>>>(w_ccam, wc);
  cvt_bf16_kernel<<<128, 256, 0, stream>>>(w_q, wqkv, 32768);
  cvt_bf16_kernel<<<128, 256, 0, stream>>>(w_k, wqkv + 32768, 32768);
  cvt_bf16_kernel<<<256, 256, 0, stream>>>(w_v, wqkv + 65536, 65536);
  copy_f32_kernel<<<1, 256, 0, stream>>>(b_q, bqkv, 128);
  copy_f32_kernel<<<1, 256, 0, stream>>>(b_k, bqkv + 128, 128);
  copy_f32_kernel<<<1, 256, 0, stream>>>(b_v, bqkv + 256, 256);
  cvt_bf16_kernel<<<16, 256, 0, stream>>>(w_enc, wencb, 4096);
  cvt_bf16_kernel<<<512, 256, 0, stream>>>(w_pw, wpwb, 131072);
  cvt_bf16_kernel<<<256, 256, 0, stream>>>(w_row, wrowb, 65536);
  cvt_bf16_kernel<<<256, 256, 0, stream>>>(w_col, wcolb, 65536);
  cvt_bf16_kernel<<<256, 256, 0, stream>>>(w_proj, wprojb, 65536);

  // ---- CCAM branch ----
  conv3x3_wmma_kernel<<<dim3(2048, 1), 512, 0, stream>>>(xb, wc, b_ccam, ccam_b, ccam_bf);
  gemm_bf16_nt<128><<<dim3(2048, 1), 512, 0, stream>>>(ccam_bf, wencb, b_enc, ccam_f,
                                                       131072, 16, 256, 1);
  energy_kernel<<<128, 256, 0, stream>>>(ccam_b, ccam_f, energy);
  ccam_softmax_kernel<<<8, 256, 0, stream>>>(energy, attnb);
  xf_kernel<<<131072, 256, 0, stream>>>(ccam_b, ccam_f, attnb, scale, xfb);

  // ---- q,k,v ----
  gemm_bf16_nt<256><<<dim3(2048, 2), 512, 0, stream>>>(xfb, wqkv, bqkv, qkvb,
                                                       131072, 512, 256, 0);

  // ---- shunted row/col tensors ----
  shunt_kernel<<<2048, 256, 0, stream>>>(qkvb, pos_rq, qs_r, 0,   128, 1, 524288);
  shunt_kernel<<<2048, 256, 0, stream>>>(qkvb, pos_rk, ks_r, 128, 128, 1, 524288);
  shunt_kernel<<<4096, 256, 0, stream>>>(qkvb, nullptr, vs_r, 256, 256, 1, 1048576);
  shunt_kernel<<<2048, 256, 0, stream>>>(qkvb, pos_cq, qs_c, 0,   128, 0, 524288);
  shunt_kernel<<<2048, 256, 0, stream>>>(qkvb, pos_ck, ks_c, 128, 128, 0, 524288);
  shunt_kernel<<<4096, 256, 0, stream>>>(qkvb, nullptr, vs_c, 256, 256, 0, 1048576);

  // ---- axial attention + 1x1 convs ----
  axial_attn_kernel<<<256, 128, 0, stream>>>(qs_r, ks_r, vs_r, xrow_a);
  axial_attn_kernel<<<256, 128, 0, stream>>>(qs_c, ks_c, vs_c, xcol_a);
  gemm_bf16_nt<256><<<dim3(64, 1), 512, 0, stream>>>(xrow_a, wrowb, b_row, xrow_o,
                                                     4096, 256, 256, 0);
  gemm_bf16_nt<256><<<dim3(64, 1), 512, 0, stream>>>(xcol_a, wcolb, b_col, xcol_o,
                                                     4096, 256, 256, 0);

  // ---- fused conv path: depthwise + pointwise ----
  dwconv_kernel<<<262144, 256, 0, stream>>>(qkvb, w_dw, b_dw, dw_a);
  gemm_bf16_nt<256><<<dim3(2048, 1), 512, 0, stream>>>(dw_a, wpwb, b_pw, pw_o,
                                                       131072, 256, 512, 0);

  // ---- combine + proj + final ----
  xx_kernel<<<131072, 256, 0, stream>>>(qkvb, xrow_o, xcol_o, xx_a);
  gemm_bf16_nt<256><<<dim3(2048, 1), 512, 0, stream>>>(xx_a, wprojb, b_proj, proj_o,
                                                       131072, 256, 256, 0);
  final_kernel<<<131072, 256, 0, stream>>>(proj_o, pw_o, out);
}
